// DeformableAttention1D_86517821211051
// MI455X (gfx1250) — compile-verified, matrix operand order checked by reading, not run on hardware
//
#include <hip/hip_runtime.h>
#include <math.h>

// ---------------------------------------------------------------------------
// DeformableAttention1D forward, MI455X (gfx1250, wave32, WMMA).
// B=1, DIM=256, N=1024, 8 heads x 32 dim_head, groups=8, m=128.
// Dominant cost: CPB bias MLP (1->64->64->1 per (i,j) pair, 8x1024x128 pairs,
// ~8.8 GFLOP) -> mapped onto v_wmma_f32_16x16x32_f16 (M=16 positions/tile).
// sim / attn@v / output projection use v_wmma_f32_16x16x4_f32.
// v-matrix staging uses gfx1250 async global->LDS copies (ASYNCcnt).
// ---------------------------------------------------------------------------

typedef __attribute__((ext_vector_type(2)))  float    v2f;
typedef __attribute__((ext_vector_type(8)))  float    v8f;
typedef __attribute__((ext_vector_type(16))) _Float16 v16h;

#define NSEQ 1024   // sequence length n
#define MKV  128    // downsampled kv length m
#define NGRP 8      // groups == heads
#define CPG  32     // channels per group == dim_head
#define CPB  64     // CPB MLP hidden dim

static __device__ __forceinline__ v8f wmma_f32x4(v2f a, v2f b, v8f c) {
  // D(16x16,f32) = A(16x4,f32) * B(4x16,f32) + C
  return __builtin_amdgcn_wmma_f32_16x16x4_f32(false, a, false, b, (short)0, c,
                                               false, false);
}
static __device__ __forceinline__ v8f wmma_f16x32(v16h a, v16h b, v8f c) {
  // D(16x16,f32) = A(16x32,f16) * B(32x16,f16) + C
  return __builtin_amdgcn_wmma_f32_16x16x32_f16(false, a, false, b, (short)0, c,
                                                false, false);
}

// async global->LDS b32 copy: lds_off = LDS byte offset (low 32 bits of a
// __shared__ generic pointer), gsrc = 64-bit global address. ASYNCcnt-tracked.
static __device__ __forceinline__ void async_g2l_b32(unsigned lds_off,
                                                     const float* gsrc) {
  asm volatile("global_load_async_to_lds_b32 %0, %1, off"
               :: "v"(lds_off), "v"(gsrc) : "memory");
}
static __device__ __forceinline__ void wait_asynccnt0() {
  asm volatile("s_wait_asynccnt 0x0" ::: "memory");
}

// ===========================================================================
// Kernel A: grouped q conv, offsets (dwconv+GELU+pwconv+tanh), kv bilinear
// sampling, grouped k/v convs. One block per group. All small vs the MLP.
// ===========================================================================
__global__ __launch_bounds__(256) void ka_prep(
    const float* __restrict__ x,      // (256, 1024)
    const float* __restrict__ w_q,    // (8, 32, 32)
    const float* __restrict__ w_k,
    const float* __restrict__ w_v,
    const float* __restrict__ w_dw,   // (32, 1, 8)
    const float* __restrict__ b_dw,   // (32)
    const float* __restrict__ w_pw,   // (32)
    float* __restrict__ q_ws,         // (256, 1024)  (pre-scaled by d^-1/2)
    float* __restrict__ k_ws,         // (256, 128)
    float* __restrict__ v_ws,         // (256, 128)
    float* __restrict__ vg_ws) {      // (8, 128) vgrid_scaled
  const int g = blockIdx.x;
  const int t = threadIdx.x;

  __shared__ float wq_s[CPG * CPG], wk_s[CPG * CPG], wv_s[CPG * CPG];
  __shared__ float wdw_s[CPG * 8], bdw_s[CPG], wpw_s[CPG];
  __shared__ float xc[CPG][128];      // x chunk (unscaled)
  __shared__ float qc[CPG][128];      // q chunk (unscaled, for offsets)
  __shared__ float offs[MKV];         // vgrid_scaled per position
  __shared__ float kv_s[CPG][MKV];    // sampled kv for this group

  for (int i = t; i < CPG * CPG; i += 256) {
    wq_s[i] = w_q[g * CPG * CPG + i];
    wk_s[i] = w_k[g * CPG * CPG + i];
    wv_s[i] = w_v[g * CPG * CPG + i];
  }
  if (t < CPG * 8) wdw_s[t] = w_dw[t];
  if (t < CPG) { bdw_s[t] = b_dw[t]; wpw_s[t] = w_pw[t]; }
  __syncthreads();

  const float scale = 0.17677669529663687f;  // 32^-0.5

  // ---- q conv + offsets, chunked over n (128 cols -> 16 offset positions)
  for (int chunk = 0; chunk < NSEQ / 128; ++chunk) {
    const int n0 = chunk * 128;
    for (int i = t; i < CPG * 128; i += 256) {
      const int c = i >> 7, n = i & 127;
      xc[c][n] = x[(g * CPG + c) * NSEQ + n0 + n];
    }
    __syncthreads();
    for (int i = t; i < CPG * 128; i += 256) {
      const int c = i >> 7, n = i & 127;
      float acc = 0.f;
      #pragma unroll 8
      for (int ci = 0; ci < CPG; ++ci) acc += wq_s[c * CPG + ci] * xc[ci][n];
      qc[c][n] = acc;                                     // unscaled for offsets
      q_ws[(g * CPG + c) * NSEQ + n0 + n] = acc * scale;  // scaled for attention
    }
    __syncthreads();
    if (t < 16) {  // 16 offset positions in this chunk
      const int p = chunk * 16 + t;
      float s = 0.f;
      for (int c = 0; c < CPG; ++c) {
        float a = bdw_s[c];
        #pragma unroll
        for (int kk = 0; kk < 8; ++kk) a += qc[c][t * 8 + kk] * wdw_s[c * 8 + kk];
        // exact GELU: 0.5*a*(1+erf(a/sqrt(2)))
        const float gl = 0.5f * a * (1.0f + erff(a * 0.7071067811865476f));
        s += gl * wpw_s[c];
      }
      const float off = tanhf(s) * 8.0f;                  // OFFSET_SCALE = 8
      const float vg = ((float)p + off) * (2.0f / 127.0f) - 1.0f;
      offs[p] = vg;
      vg_ws[g * MKV + p] = vg;
    }
    __syncthreads();
  }

  // ---- bilinear kv sampling (zeros padding, align_corners=False)
  for (int i = t; i < CPG * MKV; i += 256) {
    const int c = i >> 7, p = i & 127;
    const float vg = offs[p];
    const float pos = ((vg + 1.0f) * 1024.0f - 1.0f) * 0.5f;
    const float fi = floorf(pos);
    const int i0 = (int)fi;
    const float w = pos - fi;
    const float* xr = x + (g * CPG + c) * NSEQ;
    const int i1 = i0 + 1;
    const float v0 = (i0 >= 0 && i0 < NSEQ) ? xr[i0] : 0.f;
    const float v1 = (i1 >= 0 && i1 < NSEQ) ? xr[i1] : 0.f;
    kv_s[c][p] = v0 * (1.0f - w) + v1 * w;
  }
  __syncthreads();

  // ---- k, v grouped 1x1 convs
  for (int i = t; i < CPG * MKV; i += 256) {
    const int c = i >> 7, p = i & 127;
    float ak = 0.f, av = 0.f;
    #pragma unroll 8
    for (int ci = 0; ci < CPG; ++ci) {
      const float s = kv_s[ci][p];
      ak += wk_s[c * CPG + ci] * s;
      av += wv_s[c * CPG + ci] * s;
    }
    k_ws[(g * CPG + c) * MKV + p] = ak;
    v_ws[(g * CPG + c) * MKV + p] = av;
  }
}

// ===========================================================================
// Kernel B: fused attention. Grid = 8 heads x 16 row-blocks; block = 4 waves.
// Each wave owns a 16-row strip:
//   1) sim = q.k^T            via v_wmma_f32_16x16x4_f32   (64 WMMA)
//   2) CPB bias MLP per col   via v_wmma_f32_16x16x32_f16  (128 cols x 8 WMMA)
//   3) row softmax in LDS (exp unnormalized, 1/sum folded into epilogue)
//   4) out = attn.v           via v_wmma_f32_16x16x4_f32   (64 WMMA)
// ===========================================================================
__global__ __launch_bounds__(128) void kb_attn(
    const float* __restrict__ q_ws,   // (256,1024), pre-scaled
    const float* __restrict__ k_ws,   // (256,128)
    const float* __restrict__ v_ws,   // (256,128)
    const float* __restrict__ vg_ws,  // (8,128)
    const float* __restrict__ w1,     // (64)
    const float* __restrict__ b1,     // (64)
    const float* __restrict__ w2,     // (64,64) row-major (out,in)
    const float* __restrict__ b2,     // (64)
    const float* __restrict__ w3,     // (64)
    const float* __restrict__ b3,     // (1)
    float* __restrict__ ao_ws) {      // (256,1024) pre-projection output
  const int h   = blockIdx.x >> 4;
  const int rb  = blockIdx.x & 15;
  const int wv  = threadIdx.x >> 5;
  const int lane = threadIdx.x & 31;
  const int row0 = rb * 64 + wv * 16;          // wave's first row
  const int lmod = lane & 15;
  const int halfHi = (lane >= 16) ? 1 : 0;     // lane half (K-split for WMMA)

  __shared__ float v_lds[MKV][33];             // v[j][d], padded
  __shared__ float vg_lds[MKV];
  __shared__ float s_lds[4][16][129];          // per-wave scores, padded
  __shared__ float rsum[4][16];

  // ---- stage v into LDS with gfx1250 async global->LDS copies ----------
  // (transpose (d,j)->(j,d); per-lane LDS offset = low 32 bits of the
  //  generic shared-memory address, per the LDS aperture mapping)
  for (int i = threadIdx.x; i < MKV * CPG; i += 128) {
    const int j = i >> 5, d = i & 31;
    const unsigned loff = (unsigned)(uintptr_t)(&v_lds[j][d]);
    async_g2l_b32(loff, &v_ws[(h * CPG + d) * MKV + j]);
  }
  if (threadIdx.x < MKV) vg_lds[threadIdx.x] = vg_ws[h * MKV + threadIdx.x];
  // prefetch the k rows this block is about to stream through WMMA
  __builtin_prefetch(&k_ws[(h * CPG) * MKV], 0, 1);
  wait_asynccnt0();
  __syncthreads();

  // ---- per-lane constant preloads --------------------------------------
  // f16 A/B 16-bit layout: lane L holds row/col L%16; K indices for VGPR v:
  //   k_local(v) = (v<4?0:16) + (L>=16?8:0) + (v&3)*2  (+0/+1 packed)
  float w1r[32], b1r[32];
  #pragma unroll
  for (int kc = 0; kc < 2; ++kc)
    #pragma unroll
    for (int v = 0; v < 8; ++v) {
      const int kl = ((v < 4) ? 0 : 16) + (halfHi ? 8 : 0) + (v & 3) * 2;
      const int d = kc * 32 + kl;
      w1r[kc * 16 + 2 * v]     = w1[d];
      w1r[kc * 16 + 2 * v + 1] = w1[d + 1];
      b1r[kc * 16 + 2 * v]     = b1[d];
      b1r[kc * 16 + 2 * v + 1] = b1[d + 1];
    }
  // W2^T in B-matrix layout: B[k=d_in][n=d_out] = w2[d_out][d_in]
  v16h Bw[8];
  #pragma unroll
  for (int kc = 0; kc < 2; ++kc)
    #pragma unroll
    for (int nt = 0; nt < 4; ++nt) {
      v16h bvec;
      const int nidx = nt * 16 + lmod;
      #pragma unroll
      for (int v = 0; v < 8; ++v) {
        const int kl = ((v < 4) ? 0 : 16) + (halfHi ? 8 : 0) + (v & 3) * 2;
        const int k = kc * 32 + kl;
        bvec[2 * v]     = (_Float16)w2[nidx * CPB + k];
        bvec[2 * v + 1] = (_Float16)w2[nidx * CPB + k + 1];
      }
      Bw[kc * 4 + nt] = bvec;
    }
  float b2r[4], w3r[4];
  #pragma unroll
  for (int nt = 0; nt < 4; ++nt) {
    b2r[nt] = b2[nt * 16 + lmod];
    w3r[nt] = w3[nt * 16 + lmod];
  }
  const float b3s = b3[0];

  // q rows in f32 A layout (M=row in lanes, K split over lane halves)
  float qa[16];
  const int row_a = row0 + lmod;
  #pragma unroll
  for (int kk = 0; kk < 8; ++kk) {
    const int d = kk * 4 + halfHi * 2;
    qa[2 * kk]     = q_ws[(h * CPG + d) * NSEQ + row_a];
    qa[2 * kk + 1] = q_ws[(h * CPG + d + 1) * NSEQ + row_a];
  }
  const float seq_i = (float)row_a * (2.0f / 1023.0f) - 1.0f;

  // ---- Phase 1: sim = q.k^T  (8 col-tiles x 8 K-steps of WMMA f32 16x16x4)
  const int jcol = lmod;   // column within tile for B / D
  const int rb8 = halfHi * 8;
  for (int jt = 0; jt < 8; ++jt) {
    v8f acc = {};
    const int j = jt * 16 + jcol;
    #pragma unroll
    for (int kk = 0; kk < 8; ++kk) {
      const int d = kk * 4 + halfHi * 2;
      v2f a = {qa[2 * kk], qa[2 * kk + 1]};
      v2f b = {k_ws[(h * CPG + d) * MKV + j], k_ws[(h * CPG + d + 1) * MKV + j]};
      acc = wmma_f32x4(a, b, acc);
    }
    #pragma unroll
    for (int r = 0; r < 8; ++r) s_lds[wv][r + rb8][jt * 16 + jcol] = acc[r];
  }

  // ---- Phase 2: CPB bias MLP, one 16-row tile per kv column ------------
  for (int j = 0; j < MKV; ++j) {
    const float pd = seq_i - vg_lds[j];
    const float pos = copysignf(log1pf(fabsf(pd)), pd);
    v16h a0, a1;                         // layer-1 activations in A layout
    #pragma unroll
    for (int v = 0; v < 8; ++v) {
      a0[2 * v]     = (_Float16)fmaxf(0.f, pos * w1r[2 * v]      + b1r[2 * v]);
      a0[2 * v + 1] = (_Float16)fmaxf(0.f, pos * w1r[2 * v + 1]  + b1r[2 * v + 1]);
      a1[2 * v]     = (_Float16)fmaxf(0.f, pos * w1r[16 + 2 * v] + b1r[16 + 2 * v]);
      a1[2 * v + 1] = (_Float16)fmaxf(0.f, pos * w1r[16 + 2 * v + 1] + b1r[16 + 2 * v + 1]);
    }
    float pr[8];
    #pragma unroll
    for (int r = 0; r < 8; ++r) pr[r] = 0.f;
    #pragma unroll
    for (int nt = 0; nt < 4; ++nt) {     // layer 2: H2 = relu(A.W2^T + b2)
      v8f acc = {};
      acc = wmma_f16x32(a0, Bw[nt], acc);
      acc = wmma_f16x32(a1, Bw[4 + nt], acc);
      const float w3n = w3r[nt], b2n = b2r[nt];
      #pragma unroll
      for (int r = 0; r < 8; ++r) pr[r] += fmaxf(0.f, acc[r] + b2n) * w3n;
    }
    // layer 3: reduce hidden dim across the 16 lanes of each half-wave
    #pragma unroll
    for (int s = 1; s <= 8; s <<= 1)
      #pragma unroll
      for (int r = 0; r < 8; ++r) pr[r] += __shfl_xor(pr[r], s, 32);
    if (lmod == 0) {
      #pragma unroll
      for (int r = 0; r < 8; ++r) s_lds[wv][r + rb8][j] += pr[r] + b3s;
    }
  }

  // ---- Phase 3: row softmax (unnormalized exp; fold 1/sum into epilogue)
  {
    const int srow = lmod;
    const int c0 = halfHi * 64;
    float mx = -3.4e38f;
    for (int c = 0; c < 64; ++c) mx = fmaxf(mx, s_lds[wv][srow][c0 + c]);
    mx = fmaxf(mx, __shfl_xor(mx, 16, 32));
    float sm = 0.f;
    for (int c = 0; c < 64; ++c) {
      const float e = __expf(s_lds[wv][srow][c0 + c] - mx);
      s_lds[wv][srow][c0 + c] = e;
      sm += e;
    }
    sm += __shfl_xor(sm, 16, 32);
    if (lane < 16) rsum[wv][lane] = 1.0f / sm;
  }

  // ---- Phase 4: out = attn.v  (2 d-tiles x 32 K-steps of WMMA f32 16x16x4)
  v8f oacc0 = {}, oacc1 = {};
  for (int kk = 0; kk < 32; ++kk) {
    const int jb = kk * 4 + halfHi * 2;
    v2f a = {s_lds[wv][lmod][jb], s_lds[wv][lmod][jb + 1]};
    v2f b0 = {v_lds[jb][lmod], v_lds[jb + 1][lmod]};
    v2f b1v = {v_lds[jb][16 + lmod], v_lds[jb + 1][16 + lmod]};
    oacc0 = wmma_f32x4(a, b0, oacc0);
    oacc1 = wmma_f32x4(a, b1v, oacc1);
  }
  #pragma unroll
  for (int r = 0; r < 8; ++r) {
    const int row = row0 + r + rb8;
    const float inv = rsum[wv][r + rb8];
    ao_ws[(h * CPG + lmod) * NSEQ + row] = oacc0[r] * inv;
    ao_ws[(h * CPG + 16 + lmod) * NSEQ + row] = oacc1[r] * inv;
  }
}

// ===========================================================================
// Kernel C: y = w_out @ ao + b_out  (256x256x1024, WMMA f32 16x16x4).
// Grid = 4 o-blocks x 8 n-blocks; block = 4 waves, wave = 16 o-rows x 128 n.
// ===========================================================================
__global__ __launch_bounds__(128) void kc_proj(
    const float* __restrict__ w_out,  // (256,256)
    const float* __restrict__ b_out,  // (256)
    const float* __restrict__ ao,     // (256,1024)
    float* __restrict__ y) {          // (256,1024)
  const int ob = blockIdx.x >> 3, nb = blockIdx.x & 7;
  const int wv = threadIdx.x >> 5, lane = threadIdx.x & 31;
  const int lmod = lane & 15;
  const int halfHi = (lane >= 16) ? 1 : 0;
  const int o0 = ob * 64 + wv * 16;
  const int n0 = nb * 128;
  const int mrow = o0 + lmod;

  __builtin_prefetch(&w_out[mrow * 256], 0, 1);  // global_prefetch_b8

  v8f acc[8] = {};
  for (int kk = 0; kk < 64; ++kk) {
    const int kb = kk * 4 + halfHi * 2;
    v2f a = {w_out[mrow * 256 + kb], w_out[mrow * 256 + kb + 1]};
    #pragma unroll
    for (int nt = 0; nt < 8; ++nt) {
      const int n = n0 + nt * 16 + lmod;
      v2f b = {ao[kb * NSEQ + n], ao[(kb + 1) * NSEQ + n]};
      acc[nt] = wmma_f32x4(a, b, acc[nt]);
    }
  }
  const int rb8 = halfHi * 8;
  #pragma unroll
  for (int nt = 0; nt < 8; ++nt) {
    const int n = n0 + nt * 16 + lmod;
    #pragma unroll
    for (int r = 0; r < 8; ++r) {
      const int o = o0 + r + rb8;
      y[o * NSEQ + n] = acc[nt][r] + b_out[o];
    }
  }
}

// ===========================================================================
extern "C" void kernel_launch(void* const* d_in, const int* in_sizes, int n_in,
                              void* d_out, int out_size, void* d_ws, size_t ws_size,
                              hipStream_t stream) {
  (void)in_sizes; (void)n_in; (void)out_size; (void)ws_size;
  const float* x     = (const float*)d_in[0];
  const float* w_q   = (const float*)d_in[1];
  const float* w_k   = (const float*)d_in[2];
  const float* w_v   = (const float*)d_in[3];
  const float* w_out = (const float*)d_in[4];
  const float* b_out = (const float*)d_in[5];
  const float* w_dw  = (const float*)d_in[6];
  const float* b_dw  = (const float*)d_in[7];
  const float* w_pw  = (const float*)d_in[8];
  const float* w1    = (const float*)d_in[9];
  const float* b1    = (const float*)d_in[10];
  const float* w2    = (const float*)d_in[11];
  const float* b2    = (const float*)d_in[12];
  const float* w3    = (const float*)d_in[13];
  const float* b3    = (const float*)d_in[14];

  float* ws    = (float*)d_ws;
  float* q_ws  = ws;                          // 256*1024
  float* k_ws  = q_ws + 256 * 1024;           // 256*128
  float* v_ws  = k_ws + 256 * 128;            // 256*128
  float* vg_ws = v_ws + 256 * 128;            // 8*128
  float* ao_ws = vg_ws + 8 * 128;             // 256*1024
  float* y = (float*)d_out;

  ka_prep<<<8, 256, 0, stream>>>(x, w_q, w_k, w_v, w_dw, b_dw, w_pw,
                                 q_ws, k_ws, v_ws, vg_ws);
  kb_attn<<<128, 128, 0, stream>>>(q_ws, k_ws, v_ws, vg_ws,
                                   w1, b1, w2, b2, w3, b3, ao_ws);
  kc_proj<<<32, 128, 0, stream>>>(w_out, b_out, ao_ws, y);
}
